// BiLSTMSeq2Seq_68186900791640
// MI455X (gfx1250) — matrix operand order repaired and checked by reading, not compiled, not run on hardware
//
#include <hip/hip_runtime.h>
#include <hip/hip_bf16.h>
#include <cstdint>

#define LSq 128
#define LTq 128
#define Bq  16

typedef unsigned short u16;
typedef unsigned int   u32;
typedef __attribute__((ext_vector_type(16))) __bf16 vbf16;
typedef __attribute__((ext_vector_type(8)))  float  vf8;

union FragU { vbf16 v; uint4 q[2]; };

__device__ __forceinline__ vbf16 load_frag_split(const u16* p0, const u16* p1) {
  FragU f; f.q[0] = *(const uint4*)p0; f.q[1] = *(const uint4*)p1; return f.v;
}
__device__ __forceinline__ vbf16 load_frag_contig(const u16* p) {
  FragU f; f.q[0] = *(const uint4*)p; f.q[1] = *(const uint4*)(p + 16); return f.v;
}
__device__ __forceinline__ vf8 wmma_bf16(vbf16 a, vbf16 b, vf8 c) {
  return __builtin_amdgcn_wmma_f32_16x16x32_bf16(false, a, false, b, (short)0, c, false, false);
}
__device__ __forceinline__ u16 f2bf(float f) {
  u32 u = __float_as_uint(f);
  u32 r = u + 0x7FFFu + ((u >> 16) & 1u);
  return (u16)(r >> 16);
}
__device__ __forceinline__ float bf2f(u16 h) { return __uint_as_float(((u32)h) << 16); }
__device__ __forceinline__ vf8 splat8(float x) {
  vf8 r;
#pragma unroll
  for (int i = 0; i < 8; i++) r[i] = x;
  return r;
}
__device__ __forceinline__ float sigmoidf_(float x) { return 1.0f / (1.0f + __expf(-x)); }
__device__ __forceinline__ float wave_sum(float s) {
#pragma unroll
  for (int off = 16; off > 0; off >>= 1) s += __shfl_xor(s, off, 32);
  return s;
}
__device__ __forceinline__ float wave_max(float s) {
#pragma unroll
  for (int off = 16; off > 0; off >>= 1) s = fmaxf(s, __shfl_xor(s, off, 32));
  return s;
}

// dst[n*K + k] = bf16(src[(k + rowOff)*N + n])   (convert + transpose)
__global__ void k_cvt_transpose(const float* __restrict__ src, u16* __restrict__ dst,
                                int K, int N, int rowOff) {
  long total = (long)K * N;
  for (long i = blockIdx.x * (long)blockDim.x + threadIdx.x; i < total;
       i += (long)gridDim.x * blockDim.x) {
    int n = (int)(i / K);
    int k = (int)(i % K);
    dst[(long)n * K + k] = f2bf(src[(long)(k + rowOff) * N + n]);
  }
}

// dst[t][b][e] = bf16(emb[idx[b][t]][e])   (time-major gather)
__global__ void k_gather(const float* __restrict__ emb, const int* __restrict__ idx,
                         u16* __restrict__ dst, int L, int Bn, int E) {
  long total = (long)L * Bn * E;
  for (long i = blockIdx.x * (long)blockDim.x + threadIdx.x; i < total;
       i += (long)gridDim.x * blockDim.x) {
    int e = (int)(i % E);
    long r = i / E;
    int b = (int)(r % Bn);
    int t = (int)(r / Bn);
    dst[i] = f2bf(emb[(long)idx[b * L + t] * E + e]);
  }
}

// BiLSTM encoder: blockIdx.x = direction; 32 waves; cell state register-resident.
__global__ void __launch_bounds__(1024) k_encoder(
    const u16* __restrict__ Xenc,
    const u16* __restrict__ WihT_f, const u16* __restrict__ WhhT_f, const float* __restrict__ b_f,
    const u16* __restrict__ WihT_b, const u16* __restrict__ WhhT_b, const float* __restrict__ b_b,
    float* __restrict__ enc_out_f32, u16* __restrict__ enc_out_bf,
    float* __restrict__ dec_init_c) {
  __shared__ u16 sh_x[16 * 512];
  __shared__ u16 sh_h[16 * 512];
  const int dir = blockIdx.x;
  const u16* WihT = dir ? WihT_b : WihT_f;
  const u16* WhhT = dir ? WhhT_b : WhhT_f;
  const float* bias = dir ? b_b : b_f;
  const int tid = threadIdx.x;
  const int wave = tid >> 5, lane = tid & 31;
  const int l15 = lane & 15, kg = lane >> 4;
  const int col = 16 * wave + l15;  // h column (0..511), same for all 4 gates

  for (int i = tid; i < 16 * 512; i += 1024) sh_h[i] = 0;
  vf8 creg = splat8(0.0f);
  float bgate[4];
#pragma unroll
  for (int g = 0; g < 4; g++) bgate[g] = bias[g * 512 + col];
  __syncthreads();

  for (int s = 0; s < LSq; s++) {
    int t = dir ? (LSq - 1 - s) : s;
    {  // cooperative x_t load
      const uint4* src = (const uint4*)(Xenc + (long)t * 16 * 512);
      uint4* dq = (uint4*)sh_x;
      for (int i = tid; i < 16 * 512 / 8; i += 1024) dq[i] = src[i];
    }
    __syncthreads();
    vf8 acc[4];
#pragma unroll
    for (int g = 0; g < 4; g++) acc[g] = splat8(bgate[g]);
#pragma unroll 1
    for (int k0 = 0; k0 < 512; k0 += 32) {  // x @ Wih
      vbf16 af = load_frag_split(sh_x + (l15 * 512 + k0 + kg * 8),
                                 sh_x + (l15 * 512 + k0 + 16 + kg * 8));
#pragma unroll
      for (int g = 0; g < 4; g++) {
        vbf16 bf = load_frag_contig(WihT + ((long)(g * 512 + col)) * 512 + k0 + kg * 8);
        acc[g] = wmma_bf16(af, bf, acc[g]);
      }
    }
#pragma unroll 1
    for (int k0 = 0; k0 < 512; k0 += 32) {  // h @ Whh
      vbf16 af = load_frag_split(sh_h + (l15 * 512 + k0 + kg * 8),
                                 sh_h + (l15 * 512 + k0 + 16 + kg * 8));
#pragma unroll
      for (int g = 0; g < 4; g++) {
        vbf16 bf = load_frag_contig(WhhT + ((long)(g * 512 + col)) * 512 + k0 + kg * 8);
        acc[g] = wmma_bf16(af, bf, acc[g]);
      }
    }
    __syncthreads();  // all sh_h reads done before overwrite
    float hv[8];
#pragma unroll
    for (int v = 0; v < 8; v++) {
      float ig = sigmoidf_(acc[0][v]);
      float fg = sigmoidf_(acc[1][v]);
      float gg = tanhf(acc[2][v]);
      float og = sigmoidf_(acc[3][v]);
      float c = fg * creg[v] + ig * gg;
      creg[v] = c;
      hv[v] = og * tanhf(c);
    }
#pragma unroll
    for (int v = 0; v < 8; v++) {
      int mv = v + 8 * kg;
      u16 hb = f2bf(hv[v]);
      sh_h[mv * 512 + col] = hb;
      long oidx = ((long)mv * LSq + t) * 1024 + dir * 512 + col;
      enc_out_f32[oidx] = hv[v];
      enc_out_bf[oidx] = hb;
    }
    __syncthreads();
  }
#pragma unroll
  for (int v = 0; v < 8; v++)
    dec_init_c[(long)(v + 8 * kg) * 1024 + dir * 512 + col] = creg[v];
}

// Generic K=1024 bf16 GEMM: C[M][N] = A[M][1024] @ BT[N][1024]^T + bias (opt. ReLU)
__global__ void __launch_bounds__(512) k_gemm1024(
    const u16* __restrict__ A, const u16* __restrict__ BT,
    const float* __restrict__ bias, float* __restrict__ C, int N, int applyRelu) {
  __shared__ u16 sA[16 * 1024];
  const int tid = threadIdx.x;
  const int wave = tid >> 5, lane = tid & 31;
  const int l15 = lane & 15, kg = lane >> 4;
  const int m0 = blockIdx.x * 16;
  const int col = blockIdx.y * 256 + 16 * wave + l15;
  {
    const uint4* src = (const uint4*)(A + (long)m0 * 1024);
    uint4* dq = (uint4*)sA;
    for (int i = tid; i < 16 * 1024 / 8; i += 512) dq[i] = src[i];
  }
  __syncthreads();
  vf8 acc = splat8(bias[col]);
#pragma unroll 4
  for (int k0 = 0; k0 < 1024; k0 += 32) {
    vbf16 af = load_frag_split(sA + (l15 * 1024 + k0 + kg * 8),
                               sA + (l15 * 1024 + k0 + 16 + kg * 8));
    vbf16 bf = load_frag_contig(BT + (long)col * 1024 + k0 + kg * 8);
    acc = wmma_bf16(af, bf, acc);
  }
#pragma unroll
  for (int v = 0; v < 8; v++) {
    float val = acc[v];
    if (applyRelu) val = fmaxf(val, 0.0f);
    C[(long)(m0 + v + 8 * kg) * N + col] = val;
  }
}

// Attention decoder: single persistent workgroup; EncProj precomputed (b_attn folded in).
__global__ void __launch_bounds__(1024) k_decoder(
    const u16* __restrict__ Xdec,            // [LT][16][512] bf16
    const u16* __restrict__ WaT_h,           // [1024][1024] bf16
    const float* __restrict__ EncProj,       // [2048][1024] fp32 (enc@W_e + b_attn)
    const float* __restrict__ v_attn,        // [1024]
    const float* __restrict__ enc_f32,       // [16][128][1024]
    const u16* __restrict__ WihT_d,          // [4096][1536]
    const u16* __restrict__ WhhT_d,          // [4096][1024]
    const float* __restrict__ b_d,           // [4096]
    const float* __restrict__ dec_init_c,    // [16][1024]
    u16* __restrict__ hs_bf) {               // [16][128][1024]
  extern __shared__ char dsm[];
  u16* buf_h = (u16*)dsm;                    // 32KB: h bf16 [16][1024]
  u16* scratch = (u16*)(dsm + 32768);        // 32KB: hW bf16, later cvec bf16
  float* e_buf = (float*)(dsm + 65536);      // 8KB : scores / softmax [16][128]
  const int tid = threadIdx.x, wave = tid >> 5, lane = tid & 31;
  const int l15 = lane & 15, kg = lane >> 4;

  for (int i = tid; i < 16 * 1024; i += 1024) {  // h0 = [hf ; hb]
    int b = i >> 10, cc = i & 1023;
    int tt = (cc < 512) ? (LSq - 1) : 0;
    buf_h[i] = f2bf(enc_f32[((long)b * LSq + tt) * 1024 + cc]);
  }
  vf8 creg[2];
  const int cset0 = 16 * wave + l15;
  const int cset1 = 512 + 16 * wave + l15;
#pragma unroll
  for (int v = 0; v < 8; v++) {
    creg[0][v] = dec_init_c[(long)(v + 8 * kg) * 1024 + cset0];
    creg[1][v] = dec_init_c[(long)(v + 8 * kg) * 1024 + cset1];
  }
  __syncthreads();

  for (int t = 0; t < LTq; t++) {
    {  // phase a: hW = h @ W_h  (wave w -> N tiles 2w, 2w+1)
      int ca0 = 32 * wave + l15, ca1 = ca0 + 16;
      vf8 a0 = splat8(0.f), a1 = splat8(0.f);
#pragma unroll 1
      for (int k0 = 0; k0 < 1024; k0 += 32) {
        vbf16 af = load_frag_split(buf_h + (l15 * 1024 + k0 + kg * 8),
                                   buf_h + (l15 * 1024 + k0 + 16 + kg * 8));
        a0 = wmma_bf16(af, load_frag_contig(WaT_h + (long)ca0 * 1024 + k0 + kg * 8), a0);
        a1 = wmma_bf16(af, load_frag_contig(WaT_h + (long)ca1 * 1024 + k0 + kg * 8), a1);
      }
#pragma unroll
      for (int v = 0; v < 8; v++) {
        int mv = v + 8 * kg;
        scratch[mv * 1024 + ca0] = f2bf(a0[v]);
        scratch[mv * 1024 + ca1] = f2bf(a1[v]);
      }
    }
    __syncthreads();
    // phase b: e[b][l] = sum_k v[k]*tanh(hW[b][k] + EncProj[b*128+l][k]); 64 pairs/wave
    float ereg0 = 0.f, ereg1 = 0.f;
#pragma unroll 1
    for (int q = 0; q < 32; q++) {
      int p = wave * 64 + q;
      int b = p >> 7;
      const float* prow = EncProj + (long)p * 1024;
      float ss = 0.f;
      for (int kk = 0; kk < 32; kk++) {
        int k = lane * 32 + kk;
        ss += v_attn[k] * tanhf(bf2f(scratch[b * 1024 + k]) + prow[k]);
      }
      float tot = wave_sum(ss);
      if (q == lane) ereg0 = tot;
    }
#pragma unroll 1
    for (int q = 0; q < 32; q++) {
      int p = wave * 64 + 32 + q;
      int b = p >> 7;
      const float* prow = EncProj + (long)p * 1024;
      float ss = 0.f;
      for (int kk = 0; kk < 32; kk++) {
        int k = lane * 32 + kk;
        ss += v_attn[k] * tanhf(bf2f(scratch[b * 1024 + k]) + prow[k]);
      }
      float tot = wave_sum(ss);
      if (q == lane) ereg1 = tot;
    }
    __syncthreads();  // hW reads done
    e_buf[wave * 64 + lane] = ereg0;
    e_buf[wave * 64 + 32 + lane] = ereg1;
    __syncthreads();
    // phase c: softmax over l (one wave per batch row)
    if (wave < 16) {
      int b = wave;
      float vals[4];
#pragma unroll
      for (int j = 0; j < 4; j++) vals[j] = e_buf[b * 128 + lane + 32 * j];
      float mx = fmaxf(fmaxf(vals[0], vals[1]), fmaxf(vals[2], vals[3]));
      mx = wave_max(mx);
      float se = 0.f;
#pragma unroll
      for (int j = 0; j < 4; j++) { vals[j] = __expf(vals[j] - mx); se += vals[j]; }
      se = wave_sum(se);
      float inv = 1.0f / se;
#pragma unroll
      for (int j = 0; j < 4; j++) e_buf[b * 128 + lane + 32 * j] = vals[j] * inv;
    }
    __syncthreads();
    // phase d: cvec[b][k] = sum_l a[b][l]*enc_out[b][l][k] -> scratch (bf16)
    {
      int b = tid >> 6;
      int k0 = (tid & 63) * 16;
      float av[16];
#pragma unroll
      for (int j = 0; j < 16; j++) av[j] = 0.f;
#pragma unroll 1
      for (int l = 0; l < 128; l++) {
        float a = e_buf[b * 128 + l];
        const float4* row = (const float4*)(enc_f32 + ((long)b * LSq + l) * 1024 + k0);
#pragma unroll
        for (int j4 = 0; j4 < 4; j4++) {
          float4 r = row[j4];
          av[j4 * 4 + 0] += a * r.x; av[j4 * 4 + 1] += a * r.y;
          av[j4 * 4 + 2] += a * r.z; av[j4 * 4 + 3] += a * r.w;
        }
      }
#pragma unroll
      for (int j = 0; j < 16; j++) scratch[b * 1024 + k0 + j] = f2bf(av[j]);
    }
    __syncthreads();
    // phase e: decoder LSTM gates, K = 512 (xemb, global) + 1024 (cvec) + 1024 (h)
    // Two passes (one per column set) to halve accumulator register pressure.
    float hv[2][8];
    const u16* xrow = Xdec + (long)t * 16 * 512;
#pragma unroll 1
    for (int s2 = 0; s2 < 2; s2++) {
      const int cs = s2 ? cset1 : cset0;
      vf8 acc[4];
#pragma unroll
      for (int g = 0; g < 4; g++) acc[g] = splat8(b_d[g * 1024 + cs]);
#pragma unroll 1
      for (int k0 = 0; k0 < 2560; k0 += 32) {
        vbf16 af;
        if (k0 < 512) {
          af = load_frag_split(xrow + (l15 * 512 + k0 + kg * 8),
                               xrow + (l15 * 512 + k0 + 16 + kg * 8));
        } else if (k0 < 1536) {
          int kk = k0 - 512;
          af = load_frag_split(scratch + (l15 * 1024 + kk + kg * 8),
                               scratch + (l15 * 1024 + kk + 16 + kg * 8));
        } else {
          int kk = k0 - 1536;
          af = load_frag_split(buf_h + (l15 * 1024 + kk + kg * 8),
                               buf_h + (l15 * 1024 + kk + 16 + kg * 8));
        }
#pragma unroll
        for (int g = 0; g < 4; g++) {
          int n = g * 1024 + cs;
          vbf16 bf = (k0 < 1536)
              ? load_frag_contig(WihT_d + (long)n * 1536 + k0 + kg * 8)
              : load_frag_contig(WhhT_d + (long)n * 1024 + (k0 - 1536) + kg * 8);
          acc[g] = wmma_bf16(af, bf, acc[g]);
        }
      }
#pragma unroll
      for (int v = 0; v < 8; v++) {
        float ig = sigmoidf_(acc[0][v]);
        float fg = sigmoidf_(acc[1][v]);
        float gg = tanhf(acc[2][v]);
        float og = sigmoidf_(acc[3][v]);
        float c = fg * creg[s2][v] + ig * gg;
        creg[s2][v] = c;
        hv[s2][v] = og * tanhf(c);
      }
    }
    __syncthreads();  // buf_h / scratch reads (both passes) done before overwrite
#pragma unroll
    for (int v = 0; v < 8; v++) {
      int mv = v + 8 * kg;
      u16 hb0 = f2bf(hv[0][v]);
      u16 hb1 = f2bf(hv[1][v]);
      buf_h[mv * 1024 + cset0] = hb0;
      buf_h[mv * 1024 + cset1] = hb1;
      hs_bf[((long)mv * LTq + t) * 1024 + cset0] = hb0;
      hs_bf[((long)mv * LTq + t) * 1024 + cset1] = hb1;
    }
    __syncthreads();
  }
}

__global__ void __launch_bounds__(256) k_logsoftmax(float* __restrict__ out, int N) {
  __shared__ float red[256];
  const int row = blockIdx.x, tid = threadIdx.x;
  float* p = out + (long)row * N;
  float mx = -1e30f;
  for (int i = tid; i < N; i += 256) mx = fmaxf(mx, p[i]);
  red[tid] = mx; __syncthreads();
  for (int s = 128; s > 0; s >>= 1) { if (tid < s) red[tid] = fmaxf(red[tid], red[tid + s]); __syncthreads(); }
  mx = red[0]; __syncthreads();
  float se = 0.f;
  for (int i = tid; i < N; i += 256) se += __expf(p[i] - mx);
  red[tid] = se; __syncthreads();
  for (int s = 128; s > 0; s >>= 1) { if (tid < s) red[tid] += red[tid + s]; __syncthreads(); }
  float lse = logf(red[0]) + mx;
  __syncthreads();
  for (int i = tid; i < N; i += 256) p[i] = p[i] - lse;
}

extern "C" void kernel_launch(void* const* d_in, const int* in_sizes, int n_in,
                              void* d_out, int out_size, void* d_ws, size_t ws_size,
                              hipStream_t stream) {
  (void)in_sizes; (void)n_in; (void)out_size; (void)ws_size;
  const int*   inp     = (const int*)  d_in[0];
  const int*   tar     = (const int*)  d_in[1];
  const float* enc_emb = (const float*)d_in[2];
  const float* dec_emb = (const float*)d_in[3];
  const float* Wih_f   = (const float*)d_in[4];
  const float* Whh_f   = (const float*)d_in[5];
  const float* b_f     = (const float*)d_in[6];
  const float* Wih_b   = (const float*)d_in[7];
  const float* Whh_b   = (const float*)d_in[8];
  const float* b_b     = (const float*)d_in[9];
  const float* W_attn  = (const float*)d_in[10];
  const float* b_attn  = (const float*)d_in[11];
  const float* v_attn  = (const float*)d_in[12];
  const float* Wih_d   = (const float*)d_in[13];
  const float* Whh_d   = (const float*)d_in[14];
  const float* b_d     = (const float*)d_in[15];
  const float* W_out   = (const float*)d_in[16];
  const float* b_out   = (const float*)d_in[17];
  float* out = (float*)d_out;

  char* ws = (char*)d_ws;
  size_t off = 0;
  auto alloc = [&](size_t bytes) -> char* {
    char* p = ws + off;
    off += (bytes + 255) & ~(size_t)255;
    return p;
  };
  u16* WihT_f = (u16*)alloc((size_t)2048 * 512 * 2);
  u16* WhhT_f = (u16*)alloc((size_t)2048 * 512 * 2);
  u16* WihT_b = (u16*)alloc((size_t)2048 * 512 * 2);
  u16* WhhT_b = (u16*)alloc((size_t)2048 * 512 * 2);
  u16* WaT_h  = (u16*)alloc((size_t)1024 * 1024 * 2);
  u16* WaT_e  = (u16*)alloc((size_t)1024 * 1024 * 2);
  u16* WihT_d = (u16*)alloc((size_t)4096 * 1536 * 2);
  u16* WhhT_d = (u16*)alloc((size_t)4096 * 1024 * 2);
  u16* WoutT  = (u16*)alloc((size_t)32000 * 1024 * 2);
  u16* Xenc   = (u16*)alloc((size_t)128 * 16 * 512 * 2);
  u16* Xdec   = (u16*)alloc((size_t)128 * 16 * 512 * 2);
  float* enc_out_f32 = (float*)alloc((size_t)16 * 128 * 1024 * 4);
  u16*   enc_out_bf  = (u16*)  alloc((size_t)16 * 128 * 1024 * 2);
  float* EncProj     = (float*)alloc((size_t)2048 * 1024 * 4);
  float* dec_init_c  = (float*)alloc((size_t)16 * 1024 * 4);
  u16*   hs_bf       = (u16*)  alloc((size_t)16 * 128 * 1024 * 2);

  // 1) weight convert + transpose (bf16, [N][K])
  k_cvt_transpose<<<1024, 256, 0, stream>>>(Wih_f, WihT_f, 512, 2048, 0);
  k_cvt_transpose<<<1024, 256, 0, stream>>>(Whh_f, WhhT_f, 512, 2048, 0);
  k_cvt_transpose<<<1024, 256, 0, stream>>>(Wih_b, WihT_b, 512, 2048, 0);
  k_cvt_transpose<<<1024, 256, 0, stream>>>(Whh_b, WhhT_b, 512, 2048, 0);
  k_cvt_transpose<<<1024, 256, 0, stream>>>(W_attn, WaT_h, 1024, 1024, 0);
  k_cvt_transpose<<<1024, 256, 0, stream>>>(W_attn, WaT_e, 1024, 1024, 1024);
  k_cvt_transpose<<<4096, 256, 0, stream>>>(Wih_d, WihT_d, 1536, 4096, 0);
  k_cvt_transpose<<<4096, 256, 0, stream>>>(Whh_d, WhhT_d, 1024, 4096, 0);
  k_cvt_transpose<<<8192, 256, 0, stream>>>(W_out, WoutT, 1024, 32000, 0);

  // 2) embedding gathers (time-major bf16)
  k_gather<<<2048, 256, 0, stream>>>(enc_emb, inp, Xenc, 128, 16, 512);
  k_gather<<<2048, 256, 0, stream>>>(dec_emb, tar, Xdec, 128, 16, 512);

  // 3) BiLSTM encoder (WMMA, register-resident cell state)
  k_encoder<<<2, 1024, 0, stream>>>(Xenc, WihT_f, WhhT_f, b_f, WihT_b, WhhT_b, b_b,
                                    enc_out_f32, enc_out_bf, dec_init_c);

  // 4) EncProj = enc_out @ W_attn[H2:,:] + b_attn  (hoisted out of the decoder loop)
  k_gemm1024<<<dim3(128, 4), 512, 0, stream>>>(enc_out_bf, WaT_e, b_attn, EncProj, 1024, 0);

  // 5) attention decoder (persistent workgroup, 72KB dynamic LDS)
  k_decoder<<<1, 1024, 73728, stream>>>(Xdec, WaT_h, EncProj, v_attn, enc_out_f32,
                                        WihT_d, WhhT_d, b_d, dec_init_c, hs_bf);

  // 6) logits = relu(hs @ W_out + b_out)
  k_gemm1024<<<dim3(128, 125), 512, 0, stream>>>(hs_bf, WoutT, b_out, out, 32000, 1);

  // 7) in-place log_softmax over vocab
  k_logsoftmax<<<2048, 256, 0, stream>>>(out, 32000);
}
